// MaskedDiffusionLM_47270410060493
// MI455X (gfx1250) — compile-verified, matrix operand order checked by reading, not run on hardware
//
#include <hip/hip_runtime.h>
#include <stdint.h>

// ---------------- model dims ----------------
#define B_  4
#define S_  1024
#define D_  1024
#define H_  16
#define DH_ 64
#define F_  4096
#define L_  8
#define V_  32000
#define BS_ (B_*S_)

typedef __attribute__((ext_vector_type(16))) _Float16 v16h;
typedef __attribute__((ext_vector_type(8)))  float    v8f;
typedef __attribute__((ext_vector_type(4)))  int      v4i;

union Frag16 { v16h v; uint4 q[2]; };
union H8 { uint4 q; _Float16 h[8]; };
union H4 { uint2 q; _Float16 h[4]; };

static __device__ inline v8f wmma_f16(v16h a, v16h b, v8f c) {
  return __builtin_amdgcn_wmma_f32_16x16x32_f16(false, a, false, b, (short)0, c,
                                                false, false);
}

// ---- CDNA5 async global->LDS copy (16B per lane), ASYNCcnt-tracked ----
static __device__ inline void async_copy16(const _Float16* g, _Float16* l) {
#if __has_builtin(__builtin_amdgcn_global_load_async_to_lds_b128)
  typedef __attribute__((address_space(1))) v4i* gp_t;
  typedef __attribute__((address_space(3))) v4i* lp_t;
  __builtin_amdgcn_global_load_async_to_lds_b128(
      (gp_t)(uintptr_t)g, (lp_t)(uint32_t)(uintptr_t)l, 0, 0);
#else
  uint32_t lo = (uint32_t)(uintptr_t)l;   // LDS offset = low 32 bits
  asm volatile("global_load_async_to_lds_b128 %0, %1, off"
               :: "v"(lo), "v"(g) : "memory");
#endif
}

static __device__ inline void wait_async0() {
#if __has_builtin(__builtin_amdgcn_s_wait_asynccnt)
  __builtin_amdgcn_s_wait_asynccnt(0);
#else
  asm volatile("s_wait_asynccnt 0x0" ::: "memory");
#endif
}

// Deferred wait: leave the newest 3 async copies (next tile) in flight.
static __device__ inline void wait_async3() {
#if __has_builtin(__builtin_amdgcn_s_wait_asynccnt)
  __builtin_amdgcn_s_wait_asynccnt(3);
#else
  asm volatile("s_wait_asynccnt 0x3" ::: "memory");
#endif
}

// =====================================================================
// f32 -> f16 conversion (weights / embedding), 8 elements per thread.
// =====================================================================
__global__ __launch_bounds__(256) void cvt_f16_k(const float* __restrict__ s,
                                                 _Float16* __restrict__ d) {
  size_t i = ((size_t)blockIdx.x * 256 + threadIdx.x) * 8;
  float4 a = *(const float4*)(s + i);
  float4 b = *(const float4*)(s + i + 4);
  H8 u;
  u.h[0] = (_Float16)a.x; u.h[1] = (_Float16)a.y;
  u.h[2] = (_Float16)a.z; u.h[3] = (_Float16)a.w;
  u.h[4] = (_Float16)b.x; u.h[5] = (_Float16)b.y;
  u.h[6] = (_Float16)b.z; u.h[7] = (_Float16)b.w;
  *(uint4*)(d + i) = u.q;
}

// =====================================================================
// Embedding gather
// =====================================================================
__global__ __launch_bounds__(256) void embed_k(const int* __restrict__ ids,
                                               const float* __restrict__ emb,
                                               float* __restrict__ h) {
  int row = blockIdx.x, tid = threadIdx.x;
  int id = ids[row];
  const float4* src = (const float4*)(emb + (size_t)id * D_);
  float4* dst = (float4*)(h + (size_t)row * D_);
  dst[tid] = src[tid];
}

// =====================================================================
// RMSNorm -> f16 output (feeds WMMA GEMMs)
// =====================================================================
__global__ __launch_bounds__(256) void rmsnorm_k(const float* __restrict__ x,
                                                 const float* __restrict__ wgt,
                                                 _Float16* __restrict__ out) {
  __shared__ float red[256];
  int row = blockIdx.x, tid = threadIdx.x;
  const float* xr = x + (size_t)row * D_;
  float ss = 0.f;
  for (int i = tid; i < D_; i += 256) { float t = xr[i]; ss += t * t; }
  red[tid] = ss;
  __syncthreads();
  for (int s2 = 128; s2 > 0; s2 >>= 1) {
    if (tid < s2) red[tid] += red[tid + s2];
    __syncthreads();
  }
  float rs = rsqrtf(red[0] * (1.0f / D_) + 1e-6f);
  _Float16* orow = out + (size_t)row * D_;
  for (int i = tid; i < D_; i += 256) orow[i] = (_Float16)(wgt[i] * xr[i] * rs);
}

// =====================================================================
// WMMA GEMM: out[M,N] f32 = A[M,K]f16 * W[N,K]f16^T (+optional f32 res)
// Double-buffered LDS; staging via GLOBAL_LOAD_ASYNC_TO_LDS_B128 with a
// deferred s_wait_asynccnt<=3 so tile i+1 copies overlap tile i WMMAs.
// Block 256 thr (8 waves); tile M=128,N=64,Kstep=32; LDS stride 40 halves.
// =====================================================================
__global__ __launch_bounds__(256) void gemm_f16(const _Float16* __restrict__ A,
                                                const _Float16* __restrict__ W,
                                                float* __restrict__ out,
                                                const float* __restrict__ res,
                                                int N, int K) {
  __shared__ _Float16 ldsA[2][128 * 40];
  __shared__ _Float16 ldsB[2][64 * 40];
  const int tid  = threadIdx.x;
  const int lane = tid & 31;
  const int wv   = tid >> 5;
  const int wm   = wv & 3, wn = wv >> 2;
  const int hi   = lane >> 4;
  const int mn   = lane & 15;
  const int bM   = blockIdx.y * 128;
  const int bN   = blockIdx.x * 64;

  const v8f z = {0.f,0.f,0.f,0.f,0.f,0.f,0.f,0.f};
  v8f acc[2][2];
  acc[0][0] = z; acc[0][1] = z; acc[1][0] = z; acc[1][1] = z;

  // Per-k-step staging: 3 async b128 copies per thread.
  auto stage = [&](int buf, int k0) {
    #pragma unroll
    for (int c = 0; c < 2; ++c) {
      int u = tid + c * 256;
      int row = u >> 2, colh = (u & 3) * 8;
      async_copy16(A + (size_t)(bM + row) * K + k0 + colh,
                   &ldsA[buf][row * 40 + colh]);
    }
    int row = tid >> 2, colh = (tid & 3) * 8;
    async_copy16(W + (size_t)(bN + row) * K + k0 + colh,
                 &ldsB[buf][row * 40 + colh]);
  };

  stage(0, 0);
  int buf = 0;
  for (int k0 = 0; k0 < K; k0 += 32) {
    const bool more = (k0 + 32) < K;
    if (more) {
      stage(buf ^ 1, k0 + 32);                 // overlap with this tile's WMMAs
      if (k0 + 64 < K)                         // gfx1250 global_prefetch_b8
        __builtin_prefetch(A + (size_t)(bM + (tid & 127)) * K + k0 + 64, 0, 1);
      wait_async3();                           // drain only the older tile
    } else {
      wait_async0();
    }
    __syncthreads();                           // tile `buf` visible to all

    Frag16 a[2], b[2];
    #pragma unroll
    for (int i = 0; i < 2; ++i) {
      int r = wm * 32 + i * 16 + mn;
      a[i].q[0] = *(const uint4*)&ldsA[buf][r * 40 + hi * 8];
      a[i].q[1] = *(const uint4*)&ldsA[buf][r * 40 + hi * 8 + 16];
    }
    #pragma unroll
    for (int j = 0; j < 2; ++j) {
      int r = wn * 32 + j * 16 + mn;
      b[j].q[0] = *(const uint4*)&ldsB[buf][r * 40 + hi * 8];
      b[j].q[1] = *(const uint4*)&ldsB[buf][r * 40 + hi * 8 + 16];
    }
    #pragma unroll
    for (int i = 0; i < 2; ++i)
      #pragma unroll
      for (int j = 0; j < 2; ++j)
        acc[i][j] = wmma_f16(a[i].v, b[j].v, acc[i][j]);

    __syncthreads();                           // done reading before overwrite
    buf ^= 1;
  }

  #pragma unroll
  for (int i = 0; i < 2; ++i)
    #pragma unroll
    for (int j = 0; j < 2; ++j)
      #pragma unroll
      for (int r = 0; r < 8; ++r) {
        int mg = bM + wm * 32 + i * 16 + r + 8 * hi;
        int ng = bN + wn * 32 + j * 16 + mn;
        size_t idx = (size_t)mg * N + ng;
        float v = acc[i][j][r];
        if (res) v += res[idx];
        out[idx] = v;
      }
}

// =====================================================================
// RoPE + head pack: q,k,v f32 [B,S,D] -> f16 [B*H,S,DH]; rope on q,k.
// =====================================================================
__global__ __launch_bounds__(64) void rope_pack_k(const float* __restrict__ q,
                                                  const float* __restrict__ k,
                                                  const float* __restrict__ v,
                                                  _Float16* __restrict__ qh,
                                                  _Float16* __restrict__ kh,
                                                  _Float16* __restrict__ vh) {
  int idx = blockIdx.x;
  int s  = idx & (S_ - 1);
  int bh = idx >> 10;
  int tid = threadIdx.x;
  int h = bh & (H_ - 1), b = bh >> 4;
  size_t src = ((size_t)(b * S_ + s)) * D_ + h * DH_;
  size_t dst = ((size_t)bh * S_ + s) * DH_;
  vh[dst + tid] = (_Float16)v[src + tid];
  if (tid < 32) {
    float inv = __expf(-9.210340372f * ((float)(2 * tid) * (1.0f / 64.f)));
    float ang = (float)s * inv;
    float sn, cs;
    __sincosf(ang, &sn, &cs);
    float q1 = q[src + tid], q2 = q[src + 32 + tid];
    qh[dst + tid]      = (_Float16)(q1 * cs - q2 * sn);
    qh[dst + 32 + tid] = (_Float16)(q1 * sn + q2 * cs);
    float k1 = k[src + tid], k2 = k[src + 32 + tid];
    kh[dst + tid]      = (_Float16)(k1 * cs - k2 * sn);
    kh[dst + 32 + tid] = (_Float16)(k1 * sn + k2 * cs);
  }
}

// =====================================================================
// Fused flash attention. Grid (S/64, B*H); block 128 (4 waves).
// K tile staged with async-LDS copies; V tile transposed manually
// (pipelined loads). Output written as f16 [B,S,D] for the O-proj GEMM.
// =====================================================================
__global__ __launch_bounds__(128) void attn_k(const _Float16* __restrict__ qh,
                                              const _Float16* __restrict__ kh,
                                              const _Float16* __restrict__ vh,
                                              _Float16* __restrict__ oh) {
  __shared__ _Float16 ldsK[32 * 72];      // [key][dh]
  __shared__ _Float16 ldsVT[64 * 40];     // [dh][key]
  __shared__ _Float16 ldsP[4 * 16 * 40];  // per-wave P tile
  const int tid = threadIdx.x, lane = tid & 31, w = tid >> 5;
  const int hi = lane >> 4, mn = lane & 15;
  const int bh = blockIdx.y, qt = blockIdx.x;
  const int b = bh >> 4, h = bh & 15;
  const float scale = 0.125f;

  int qrow = qt * 64 + w * 16 + mn;
  const _Float16* qb = qh + ((size_t)bh * S_ + qrow) * DH_;
  Frag16 qf[2];
  #pragma unroll
  for (int c = 0; c < 2; ++c) {
    qf[c].q[0] = *(const uint4*)(qb + c * 32 + hi * 8);
    qf[c].q[1] = *(const uint4*)(qb + c * 32 + hi * 8 + 16);
  }

  const v8f z = {0.f,0.f,0.f,0.f,0.f,0.f,0.f,0.f};
  float mstate[8], lstate[8];
  v8f oacc[4];
  #pragma unroll
  for (int r = 0; r < 8; ++r) { mstate[r] = -3.0e38f; lstate[r] = 0.f; }
  #pragma unroll
  for (int nt = 0; nt < 4; ++nt) oacc[nt] = z;

  _Float16* myP = &ldsP[w * 16 * 40];

  for (int kt0 = 0; kt0 < S_; kt0 += 32) {
    __syncthreads();
    // K tile: 256 x 16B chunks, async, 2 per thread.
    #pragma unroll
    for (int i = 0; i < 2; ++i) {
      int u = tid + i * 128;
      int row = u >> 3, colh = (u & 7) * 8;
      async_copy16(kh + ((size_t)bh * S_ + kt0 + row) * DH_ + colh,
                   &ldsK[row * 72 + colh]);
    }
    // V tile, transposed: pipelined (both loads in flight, then stores).
    H8 va[2];
    int vrow[2], vcol[2];
    #pragma unroll
    for (int i = 0; i < 2; ++i) {
      int u = tid + i * 128;
      vrow[i] = u >> 3; vcol[i] = (u & 7) * 8;
      va[i].q = *(const uint4*)(vh + ((size_t)bh * S_ + kt0 + vrow[i]) * DH_ +
                                vcol[i]);
    }
    #pragma unroll
    for (int i = 0; i < 2; ++i)
      #pragma unroll
      for (int e = 0; e < 8; ++e)
        ldsVT[(vcol[i] + e) * 40 + vrow[i]] = va[i].h[e];
    wait_async0();
    __syncthreads();

    v8f s0 = z, s1 = z;
    #pragma unroll
    for (int c = 0; c < 2; ++c) {
      Frag16 bf;
      int key0 = mn;
      bf.q[0] = *(const uint4*)&ldsK[key0 * 72 + c * 32 + hi * 8];
      bf.q[1] = *(const uint4*)&ldsK[key0 * 72 + c * 32 + hi * 8 + 16];
      s0 = wmma_f16(qf[c].v, bf.v, s0);
      int key1 = 16 + mn;
      bf.q[0] = *(const uint4*)&ldsK[key1 * 72 + c * 32 + hi * 8];
      bf.q[1] = *(const uint4*)&ldsK[key1 * 72 + c * 32 + hi * 8 + 16];
      s1 = wmma_f16(qf[c].v, bf.v, s1);
    }

    #pragma unroll
    for (int r = 0; r < 8; ++r) {
      float a0 = s0[r] * scale, a1 = s1[r] * scale;
      float mx = fmaxf(a0, a1);
      #pragma unroll
      for (int off = 1; off < 16; off <<= 1)
        mx = fmaxf(mx, __shfl_xor(mx, off, 32));
      float mnew = fmaxf(mstate[r], mx);
      float alpha = __expf(mstate[r] - mnew);
      float p0 = __expf(a0 - mnew), p1 = __expf(a1 - mnew);
      float ps = p0 + p1;
      #pragma unroll
      for (int off = 1; off < 16; off <<= 1)
        ps += __shfl_xor(ps, off, 32);
      lstate[r] = lstate[r] * alpha + ps;
      mstate[r] = mnew;
      int prow = r + 8 * hi;
      myP[prow * 40 + mn]      = (_Float16)p0;
      myP[prow * 40 + 16 + mn] = (_Float16)p1;
      #pragma unroll
      for (int nt = 0; nt < 4; ++nt) oacc[nt][r] *= alpha;
    }

    Frag16 pf;
    pf.q[0] = *(const uint4*)&myP[mn * 40 + hi * 8];
    pf.q[1] = *(const uint4*)&myP[mn * 40 + hi * 8 + 16];
    #pragma unroll
    for (int nt = 0; nt < 4; ++nt) {
      Frag16 vf;
      int dh = nt * 16 + mn;
      vf.q[0] = *(const uint4*)&ldsVT[dh * 40 + hi * 8];
      vf.q[1] = *(const uint4*)&ldsVT[dh * 40 + hi * 8 + 16];
      oacc[nt] = wmma_f16(pf.v, vf.v, oacc[nt]);
    }
  }

  #pragma unroll
  for (int nt = 0; nt < 4; ++nt)
    #pragma unroll
    for (int r = 0; r < 8; ++r) {
      int srow = qt * 64 + w * 16 + r + 8 * hi;
      int col = h * DH_ + nt * 16 + mn;
      oh[((size_t)(b * S_ + srow)) * D_ + col] =
          (_Float16)(oacc[nt][r] / lstate[r]);
    }
}

// =====================================================================
// x = silu(g) * u  (f32 in, f16 out, feeds the down-proj GEMM)
// =====================================================================
__global__ __launch_bounds__(256) void silu_mul_k(const float* __restrict__ g,
                                                  const float* __restrict__ u,
                                                  _Float16* __restrict__ x) {
  size_t i = ((size_t)blockIdx.x * 256 + threadIdx.x) * 4;
  float4 gv = *(const float4*)(g + i);
  float4 uv = *(const float4*)(u + i);
  H4 o;
  o.h[0] = (_Float16)(gv.x / (1.f + __expf(-gv.x)) * uv.x);
  o.h[1] = (_Float16)(gv.y / (1.f + __expf(-gv.y)) * uv.y);
  o.h[2] = (_Float16)(gv.z / (1.f + __expf(-gv.z)) * uv.z);
  o.h[3] = (_Float16)(gv.w / (1.f + __expf(-gv.w)) * uv.w);
  *(uint2*)(x + i) = o.q;
}

// =====================================================================
// Host orchestration
// =====================================================================
extern "C" void kernel_launch(void* const* d_in, const int* in_sizes, int n_in,
                              void* d_out, int out_size, void* d_ws,
                              size_t ws_size, hipStream_t stream) {
  (void)in_sizes; (void)n_in; (void)out_size; (void)ws_size;
  const int*   ids = (const int*)d_in[0];
  const float* emb = (const float*)d_in[1];
  const float* wq  = (const float*)d_in[2];
  const float* wk  = (const float*)d_in[3];
  const float* wv  = (const float*)d_in[4];
  const float* wo  = (const float*)d_in[5];
  const float* wg  = (const float*)d_in[6];
  const float* wu  = (const float*)d_in[7];
  const float* wd  = (const float*)d_in[8];
  const float* ln1 = (const float*)d_in[9];
  const float* ln2 = (const float*)d_in[10];
  const float* lnf = (const float*)d_in[11];

  char* ws = (char*)d_ws;
  size_t off = 0;
  auto carve = [&](size_t bytes) -> char* {
    char* p = ws + off;
    off += (bytes + 255) & ~(size_t)255;
    return p;
  };
  const size_t f32BD = (size_t)BS_ * D_ * 4;
  const size_t f16BD = (size_t)BS_ * D_ * 2;
  const size_t f32BF = (size_t)BS_ * F_ * 4;
  const size_t f16BF = (size_t)BS_ * F_ * 2;

  float*    hA    = (float*)carve(f32BD);
  float*    hB    = (float*)carve(f32BD);
  float*    qf    = (float*)carve(f32BD);
  float*    kf    = (float*)carve(f32BD);
  float*    vf    = (float*)carve(f32BD);
  float*    gbuf  = (float*)carve(f32BF);
  float*    ubuf  = (float*)carve(f32BF);
  _Float16* nrm16 = (_Float16*)carve(f16BD);
  _Float16* qhh   = (_Float16*)carve(f16BD);
  _Float16* khh   = (_Float16*)carve(f16BD);
  _Float16* vhh   = (_Float16*)carve(f16BD);
  _Float16* oh    = (_Float16*)carve(f16BD);
  _Float16* xh    = (_Float16*)carve(f16BF);
  _Float16* wq16  = (_Float16*)carve((size_t)D_ * D_ * 2);
  _Float16* wk16  = (_Float16*)carve((size_t)D_ * D_ * 2);
  _Float16* wv16  = (_Float16*)carve((size_t)D_ * D_ * 2);
  _Float16* wo16  = (_Float16*)carve((size_t)D_ * D_ * 2);
  _Float16* wg16  = (_Float16*)carve((size_t)F_ * D_ * 2);
  _Float16* wu16  = (_Float16*)carve((size_t)F_ * D_ * 2);
  _Float16* wd16  = (_Float16*)carve((size_t)D_ * F_ * 2);
  _Float16* emb16 = (_Float16*)carve((size_t)V_ * D_ * 2);

  const int cDD = (D_ * D_) / 2048;
  const int cFD = (F_ * D_) / 2048;
  const int cVD = (V_ * D_) / 2048;

  dim3 gD(D_ / 64, BS_ / 128);
  dim3 gF(F_ / 64, BS_ / 128);
  dim3 gV(V_ / 64, BS_ / 128);
  dim3 gAttn(S_ / 64, B_ * H_);

  cvt_f16_k<<<cVD, 256, 0, stream>>>(emb, emb16);
  embed_k<<<BS_, 256, 0, stream>>>(ids, emb, hA);

  for (int l = 0; l < L_; ++l) {
    cvt_f16_k<<<cDD, 256, 0, stream>>>(wq + (size_t)l * D_ * D_, wq16);
    cvt_f16_k<<<cDD, 256, 0, stream>>>(wk + (size_t)l * D_ * D_, wk16);
    cvt_f16_k<<<cDD, 256, 0, stream>>>(wv + (size_t)l * D_ * D_, wv16);
    cvt_f16_k<<<cDD, 256, 0, stream>>>(wo + (size_t)l * D_ * D_, wo16);
    cvt_f16_k<<<cFD, 256, 0, stream>>>(wg + (size_t)l * F_ * D_, wg16);
    cvt_f16_k<<<cFD, 256, 0, stream>>>(wu + (size_t)l * F_ * D_, wu16);
    cvt_f16_k<<<cFD, 256, 0, stream>>>(wd + (size_t)l * D_ * F_, wd16);

    rmsnorm_k<<<BS_, 256, 0, stream>>>(hA, ln1 + (size_t)l * D_, nrm16);
    gemm_f16<<<gD, 256, 0, stream>>>(nrm16, wq16, qf, nullptr, D_, D_);
    gemm_f16<<<gD, 256, 0, stream>>>(nrm16, wk16, kf, nullptr, D_, D_);
    gemm_f16<<<gD, 256, 0, stream>>>(nrm16, wv16, vf, nullptr, D_, D_);
    rope_pack_k<<<B_ * H_ * S_, 64, 0, stream>>>(qf, kf, vf, qhh, khh, vhh);
    attn_k<<<gAttn, 128, 0, stream>>>(qhh, khh, vhh, oh);
    gemm_f16<<<gD, 256, 0, stream>>>(oh, wo16, hB, hA, D_, D_);
    rmsnorm_k<<<BS_, 256, 0, stream>>>(hB, ln2 + (size_t)l * D_, nrm16);
    gemm_f16<<<gF, 256, 0, stream>>>(nrm16, wg16, gbuf, nullptr, F_, D_);
    gemm_f16<<<gF, 256, 0, stream>>>(nrm16, wu16, ubuf, nullptr, F_, D_);
    silu_mul_k<<<(BS_ * (F_ / 1024)), 256, 0, stream>>>(gbuf, ubuf, xh);
    gemm_f16<<<gD, 256, 0, stream>>>(xh, wd16, hA, hB, D_, F_);
  }

  rmsnorm_k<<<BS_, 256, 0, stream>>>(hA, lnf, nrm16);
  gemm_f16<<<gV, 256, 0, stream>>>(nrm16, emb16, (float*)d_out, nullptr, V_, D_);
}
